// SparseGlobalBlock_25494925869231
// MI455X (gfx1250) — matrix operand order repaired and checked by reading, not compiled, not run on hardware
//
#include <hip/hip_runtime.h>
#include <hip/hip_bf16.h>
#include <stdint.h>
#include <math.h>

// Problem constants (from reference setup_inputs)
#define B_SZ   16
#define C_SZ   512
#define H_SZ   48
#define W_SZ   48
#define N_SZ   (H_SZ * W_SZ)   // 2304
#define KTOP   256
#define LN_EPS 1e-5f

// ---------------------------------------------------------------------------
// CDNA5 WMMA helpers (wave32, 16x16x32 bf16 -> f32)
// ---------------------------------------------------------------------------
typedef __attribute__((ext_vector_type(16))) __bf16       v16bf;
typedef __attribute__((ext_vector_type(8)))  float        v8f;
typedef __attribute__((ext_vector_type(8)))  unsigned int v8u;

__device__ __forceinline__ v8f wmma_bf16(v16bf a, v16bf b, v8f c) {
    // (neg_a, A, neg_b, B, c_mod, C, reuse_a, reuse_b)
    return __builtin_amdgcn_wmma_f32_16x16x32_bf16(false, a, false, b,
                                                   (short)0, c, false, false);
}

__device__ __forceinline__ v8f v8f_zero() {
    v8f z = {0.f, 0.f, 0.f, 0.f, 0.f, 0.f, 0.f, 0.f};
    return z;
}

// pack two f32 -> one dword of two bf16 (lowers to v_cvt_pk_bf16_f32)
__device__ __forceinline__ unsigned pk_bf16(float a, float b) {
    union { __bf16 h[2]; unsigned u; } x;
    x.h[0] = (__bf16)a; x.h[1] = (__bf16)b;
    return x.u;
}

// Build a 16-element bf16 fragment from two aligned 16B LDS chunks
// (each lowers to ds_load_b128).
__device__ __forceinline__ v16bf frag_from_lds(const __bf16* p0, const __bf16* p1) {
    uint4 lo = *(const uint4*)p0;
    uint4 hi = *(const uint4*)p1;
    union { v8u u; v16bf b; } f;
    f.u[0] = lo.x; f.u[1] = lo.y; f.u[2] = lo.z; f.u[3] = lo.w;
    f.u[4] = hi.x; f.u[5] = hi.y; f.u[6] = hi.z; f.u[7] = hi.w;
    return f.b;
}

// Low 32 bits of a flat shared-aperture address == LDS byte offset (ISA 10.2)
__device__ __forceinline__ unsigned lds_off_of(const void* p) {
    return (unsigned)(uint64_t)(uintptr_t)p;
}

// Async global->LDS copy, GVS mode (SADDR base + 32-bit VGPR byte offset).
// Tracked with ASYNCcnt (CDNA5 ISA).
__device__ __forceinline__ void async_copy_b32(const void* sbase,
                                               unsigned gByteOff,
                                               unsigned ldsByteOff) {
    asm volatile("global_load_async_to_lds_b32 %0, %1, %2"
                 :
                 : "v"(ldsByteOff), "v"(gByteOff), "s"(sbase)
                 : "memory");
}

__device__ __forceinline__ void wait_async_all() {
    asm volatile("s_wait_asynccnt 0x0" ::: "memory");
}

// ---------------------------------------------------------------------------
// Kernel 1: importance[b][n] = sum_c x[b][c][n]^2  (nan_to_num-guarded)
// ---------------------------------------------------------------------------
__global__ void __launch_bounds__(256)
importance_kernel(const float* __restrict__ x, float* __restrict__ imp) {
    int e = blockIdx.x * 256 + threadIdx.x;      // b*N + n
    if (e >= B_SZ * N_SZ) return;
    int b = e / N_SZ;
    int n = e - b * N_SZ;
    const float* p = x + (long long)b * C_SZ * N_SZ + n;
    float s = 0.f;
    for (int c = 0; c < C_SZ; ++c) {
        float v = p[(long long)c * N_SZ];
        s += v * v;
    }
    if (!__builtin_isfinite(s)) s = 0.f;
    imp[e] = s;
}

// ---------------------------------------------------------------------------
// Kernel 2: exact top-k by rank (descending, stable ties == jax.lax.top_k)
// ---------------------------------------------------------------------------
__global__ void __launch_bounds__(256)
topk_kernel(const float* __restrict__ imp, int* __restrict__ idxOut) {
    __shared__ float simp[N_SZ];
    const int b = blockIdx.x, t = threadIdx.x;
    for (int i = t; i < N_SZ; i += 256) simp[i] = imp[b * N_SZ + i];
    __syncthreads();
    for (int n = t; n < N_SZ; n += 256) {
        float vn = simp[n];
        int rank = 0;
        for (int m = 0; m < N_SZ; ++m) {
            float vm = simp[m];
            rank += (vm > vn) || (vm == vn && m < n);
        }
        if (rank < KTOP) idxOut[b * KTOP + rank] = n;
    }
}

// ---------------------------------------------------------------------------
// Kernel 3: generic batched WMMA GEMM  O[b] = A (Co x Ci) @ B[b] (Ci x Ncols)
//   - optional column gather (idxBase): B column j -> global column idx[j]
//   - outT==1: store transposed as O[j*Co + m]   (for q_sel/k_sel (b,j,c))
//   - outT==0: store O[m*ldB + n] (+ optional residual)  (for v and final y)
// Block tile 128x128, K-step 32, 8 waves each owning a 64x32 sub-tile.
// f32 tiles staged via global_load_async_to_lds_b32 (ASYNCcnt), converted
// once to bf16 LDS tiles in frag-native layouts:
//   Abf[row][k]  : A-frag = 2x ds_load_b128 (K groups of 8 contiguous)
//   Bbf[col][k]  : B-frag = 2x ds_load_b128 (K groups of 8 contiguous)
// ---------------------------------------------------------------------------
__global__ void __launch_bounds__(256)
gemm_wmma_kernel(const float* __restrict__ A,
                 const float* __restrict__ Bbase, long long strideB,
                 float*       __restrict__ Obase, long long strideO,
                 const int*   __restrict__ idxBase, int idxStride,
                 const float* __restrict__ Rbase, long long strideR,
                 int Ci, int Co, int Ncols, int ldB, int outT) {
    __shared__ __align__(16) float  As[128 * 32];   // f32 stage [row][k]
    __shared__ __align__(16) float  Bs[32 * 128];   // f32 stage [k][col]
    __shared__ __align__(16) __bf16 Abf[128 * 32];  // bf16 [row][k]
    __shared__ __align__(16) __bf16 Bbf[128 * 32];  // bf16 [col][k] (transposed)

    const int t    = threadIdx.x;
    const int lane = t & 31;
    const int wave = t >> 5;
    const int half = lane >> 4;
    const int lrow = lane & 15;
    const int wm   = wave >> 2;      // 0..1 -> row offset wm*64
    const int wn   = wave & 3;       // 0..3 -> col offset wn*32

    const int b  = blockIdx.z;
    const int n0 = blockIdx.x * 128;
    const int m0 = blockIdx.y * 128;

    const float* Bm = Bbase + (long long)b * strideB;
    float*       Om = Obase + (long long)b * strideO;
    const float* Rm = Rbase ? (Rbase + (long long)b * strideR) : nullptr;
    const int*   ci = idxBase ? (idxBase + b * idxStride) : nullptr;

    __builtin_prefetch(A + (long long)m0 * Ci, 0, 3);   // global_prefetch_b8

    const unsigned aLds = lds_off_of(As);
    const unsigned bLds = lds_off_of(Bs);

    // K-invariant staging coordinates
    int ar[16], ak[16], bk[16], gcol[16];
#pragma unroll
    for (int i = 0; i < 16; ++i) {
        int ea = t * 16 + i;                 // A tile element (128x32)
        ar[i] = ea >> 5; ak[i] = ea & 31;
        int eb = t + i * 256;                // B tile element (32x128), coalesced
        bk[i] = eb >> 7;
        int cc = eb & 127;
        gcol[i] = ci ? ci[n0 + cc] : (n0 + cc);
    }

    // convert-pass coordinates (K-invariant)
    const int carow = t >> 1, cakb = (t & 1) * 16;          // A: 16 elems/thread
    const int cbk4  = (t >> 5) * 4, cbc4 = (t & 31) * 4;    // B: 4x4 block/thread

    v8f acc[4][2];
#pragma unroll
    for (int rt = 0; rt < 4; ++rt)
#pragma unroll
        for (int ct = 0; ct < 2; ++ct) acc[rt][ct] = v8f_zero();

    for (int k0 = 0; k0 < Ci; k0 += 32) {
        // ---- async f32 staging ----
#pragma unroll
        for (int i = 0; i < 16; ++i) {
            unsigned ga = (unsigned)(((m0 + ar[i]) * Ci + k0 + ak[i]) * 4);
            async_copy_b32(A, ga, aLds + (unsigned)((t * 16 + i) * 4));
        }
#pragma unroll
        for (int i = 0; i < 16; ++i) {
            unsigned gb = (unsigned)(((k0 + bk[i]) * ldB + gcol[i]) * 4);
            async_copy_b32(Bm, gb, bLds + (unsigned)((t + i * 256) * 4));
        }
        wait_async_all();
        __syncthreads();                     // f32 tiles visible

        // ---- one-shot f32 -> bf16 conversion ----
        {   // A: contiguous 16 floats -> 16 bf16 (2x ds_store_b128)
            const float* s = &As[carow * 32 + cakb];
            float4 f0 = *(const float4*)(s);
            float4 f1 = *(const float4*)(s + 4);
            float4 f2 = *(const float4*)(s + 8);
            float4 f3 = *(const float4*)(s + 12);
            uint4 u0, u1;
            u0.x = pk_bf16(f0.x, f0.y); u0.y = pk_bf16(f0.z, f0.w);
            u0.z = pk_bf16(f1.x, f1.y); u0.w = pk_bf16(f1.z, f1.w);
            u1.x = pk_bf16(f2.x, f2.y); u1.y = pk_bf16(f2.z, f2.w);
            u1.z = pk_bf16(f3.x, f3.y); u1.w = pk_bf16(f3.z, f3.w);
            uint4* d = (uint4*)&Abf[carow * 32 + cakb];
            d[0] = u0; d[1] = u1;
        }
        {   // B: 4x4 register transpose -> Bbf[col][k]
            float4 r0 = *(const float4*)&Bs[(cbk4 + 0) * 128 + cbc4];
            float4 r1 = *(const float4*)&Bs[(cbk4 + 1) * 128 + cbc4];
            float4 r2 = *(const float4*)&Bs[(cbk4 + 2) * 128 + cbc4];
            float4 r3 = *(const float4*)&Bs[(cbk4 + 3) * 128 + cbc4];
            float a0[4] = {r0.x, r0.y, r0.z, r0.w};
            float a1[4] = {r1.x, r1.y, r1.z, r1.w};
            float a2[4] = {r2.x, r2.y, r2.z, r2.w};
            float a3[4] = {r3.x, r3.y, r3.z, r3.w};
#pragma unroll
            for (int i = 0; i < 4; ++i) {
                uint2 u;
                u.x = pk_bf16(a0[i], a1[i]);
                u.y = pk_bf16(a2[i], a3[i]);
                *(uint2*)&Bbf[(cbc4 + i) * 32 + cbk4] = u;
            }
        }
        __syncthreads();                     // bf16 tiles ready

        // ---- fragments: 2x ds_load_b128 each; then 8 WMMAs ----
        v16bf af[4], bfr[2];
#pragma unroll
        for (int rt = 0; rt < 4; ++rt) {
            const __bf16* p = &Abf[(wm * 64 + rt * 16 + lrow) * 32 + half * 8];
            af[rt] = frag_from_lds(p, p + 16);
        }
#pragma unroll
        for (int ct = 0; ct < 2; ++ct) {
            const __bf16* p = &Bbf[(wn * 32 + ct * 16 + lrow) * 32 + half * 8];
            bfr[ct] = frag_from_lds(p, p + 16);
        }
#pragma unroll
        for (int rt = 0; rt < 4; ++rt)
#pragma unroll
            for (int ct = 0; ct < 2; ++ct)
                acc[rt][ct] = wmma_bf16(af[rt], bfr[ct], acc[rt][ct]);
        // no trailing barrier needed: next iter's post-wait barrier orders
        // f32-stage overwrites; bf16 tiles are only rewritten after it.
    }

    // Store: D layout f32 16x16 -> row = r + 8*half, col = lane&15
#pragma unroll
    for (int rt = 0; rt < 4; ++rt) {
#pragma unroll
        for (int ct = 0; ct < 2; ++ct) {
#pragma unroll
            for (int r = 0; r < 8; ++r) {
                int row = m0 + wm * 64 + rt * 16 + r + half * 8;
                int col = n0 + wn * 32 + ct * 16 + lrow;
                float val = acc[rt][ct][r];
                if (outT) {
                    Om[(long long)col * Co + row] = val;
                } else {
                    long long o = (long long)row * ldB + col;
                    if (Rm) val += Rm[o];
                    Om[o] = val;
                }
            }
        }
    }
    (void)Ncols;
}

// ---------------------------------------------------------------------------
// Kernel 4: row layernorm over C=512 (one block per row, 2 elems/thread)
// ---------------------------------------------------------------------------
__global__ void __launch_bounds__(256)
layernorm_kernel(float* __restrict__ buf, const float* __restrict__ w,
                 const float* __restrict__ bia) {
    __shared__ float rs[256], rq[256];
    const int t = threadIdx.x;
    float* p = buf + (long long)blockIdx.x * C_SZ;
    float v0 = p[t], v1 = p[t + 256];
    rs[t] = v0 + v1;
    rq[t] = v0 * v0 + v1 * v1;
    __syncthreads();
    for (int s = 128; s > 0; s >>= 1) {
        if (t < s) { rs[t] += rs[t + s]; rq[t] += rq[t + s]; }
        __syncthreads();
    }
    float mean = rs[0] * (1.0f / C_SZ);
    float var  = rq[0] * (1.0f / C_SZ) - mean * mean;
    float inv  = rsqrtf(var + LN_EPS);
    p[t]       = (v0 - mean) * inv * w[t] + bia[t];
    p[t + 256] = (v1 - mean) * inv * w[t + 256] + bia[t + 256];
}

// ---------------------------------------------------------------------------
// Kernel 5: gather v_sel[b][j][c] = v[b][c][idx[b][j]]  (avoids RW race
// between the attention scatter and its own V reads)
// ---------------------------------------------------------------------------
__global__ void __launch_bounds__(256)
gather_vsel_kernel(const float* __restrict__ v, const int* __restrict__ idx,
                   float* __restrict__ vsel) {
    long long e = (long long)blockIdx.x * 256 + threadIdx.x; // (b*K + j)*C + c
    int c  = (int)(e & (C_SZ - 1));
    long long bj = e >> 9;                 // /C_SZ
    int j  = (int)(bj & (KTOP - 1));
    int b  = (int)(bj >> 8);               // /KTOP
    int col = idx[b * KTOP + j];
    vsel[e] = v[(long long)b * C_SZ * N_SZ + (long long)c * N_SZ + col];
}

// ---------------------------------------------------------------------------
// Kernel 6: fused attention per (batch, 32 query rows):
//   S = qn @ kn^T * scale  (WMMA, K=512) -> clip/softmax in LDS ->
//   attended = P @ V_sel   (WMMA, K=256) -> scatter into vbuf columns idx[j]
// All bf16 LDS tiles use frag-native layouts: each operand = 2x ds_load_b128.
// ---------------------------------------------------------------------------
__global__ void __launch_bounds__(256)
attn_wmma_kernel(const float* __restrict__ qsel, const float* __restrict__ ksel,
                 const float* __restrict__ vsel, const int* __restrict__ idx,
                 float* __restrict__ vbuf, float scale) {
    __shared__ __align__(16) unsigned char smA[18 * 1024]; // qs+kts | pbuf
    __shared__ __align__(16) unsigned char smB[32 * 1024]; // sbuf | vs
    __bf16* qs   = (__bf16*)smA;             // [32][32]   row-major
    __bf16* kts  = (__bf16*)(smA + 2048);    // [256][32]  == k_sel layout
    __bf16* pbuf = (__bf16*)smA;             // [32][256]  (reuses qs/kts)
    float*  sbuf = (float*)smB;              // [32][256]
    __bf16* vs   = (__bf16*)smB;             // [512][32]  col-major (reuses sbuf)

    const int t = threadIdx.x, lane = t & 31, wave = t >> 5;
    const int half = lane >> 4, lrow = lane & 15;
    const int j0 = blockIdx.x * 32;
    const int b  = blockIdx.y;

    const float* qb = qsel + ((long long)b * KTOP + j0) * C_SZ;
    const float* kb = ksel + (long long)b * KTOP * C_SZ;
    const float* vb = vsel + (long long)b * KTOP * C_SZ;
    const int*   ib = idx + b * KTOP;

    // ---- phase 1: S = q @ k^T, 32x256 per block; wave owns 32 cols ----
    v8f accS[2][2];
#pragma unroll
    for (int rt = 0; rt < 2; ++rt)
#pragma unroll
        for (int cc = 0; cc < 2; ++cc) accS[rt][cc] = v8f_zero();

    for (int k0 = 0; k0 < C_SZ; k0 += 32) {
        {   // qs: 32x32, 4 contiguous elems/thread
            int e = t * 4;
            int r = e >> 5, kk = e & 31;
            float4 f = *(const float4*)&qb[(long long)r * C_SZ + k0 + kk];
            uint2 u; u.x = pk_bf16(f.x, f.y); u.y = pk_bf16(f.z, f.w);
            *(uint2*)&qs[e] = u;
        }
#pragma unroll
        for (int i = 0; i < 8; ++i) {        // kts: 256x32, contiguous both sides
            int e = t * 4 + i * 1024;
            int jj = e >> 5, kk = e & 31;
            float4 f = *(const float4*)&kb[(long long)jj * C_SZ + k0 + kk];
            uint2 u; u.x = pk_bf16(f.x, f.y); u.y = pk_bf16(f.z, f.w);
            *(uint2*)&kts[e] = u;
        }
        __syncthreads();

        v16bf af[2], bfr[2];
#pragma unroll
        for (int rt = 0; rt < 2; ++rt) {
            const __bf16* p = &qs[(rt * 16 + lrow) * 32 + half * 8];
            af[rt] = frag_from_lds(p, p + 16);
        }
#pragma unroll
        for (int cc = 0; cc < 2; ++cc) {
            const __bf16* p = &kts[(wave * 32 + cc * 16 + lrow) * 32 + half * 8];
            bfr[cc] = frag_from_lds(p, p + 16);
        }
#pragma unroll
        for (int rt = 0; rt < 2; ++rt)
#pragma unroll
            for (int cc = 0; cc < 2; ++cc)
                accS[rt][cc] = wmma_bf16(af[rt], bfr[cc], accS[rt][cc]);
        __syncthreads();
    }

    // scale / nan_to_num / clip -> sbuf
#pragma unroll
    for (int rt = 0; rt < 2; ++rt)
#pragma unroll
        for (int cc = 0; cc < 2; ++cc)
#pragma unroll
            for (int r = 0; r < 8; ++r) {
                int row = rt * 16 + r + half * 8;
                int col = wave * 32 + cc * 16 + lrow;
                float s = accS[rt][cc][r] * scale;
                if (!__builtin_isfinite(s)) s = 0.f;
                s = fminf(80.f, fmaxf(-80.f, s));
                sbuf[row * 256 + col] = s;
            }
    __syncthreads();

    // softmax per row (threads 0..31)
    if (t < 32) {
        float* rp = sbuf + t * 256;
        float m = -3.4e38f;
        for (int c = 0; c < 256; ++c) m = fmaxf(m, rp[c]);
        float sum = 0.f;
        for (int c = 0; c < 256; ++c) { float e = __expf(rp[c] - m); rp[c] = e; sum += e; }
        float inv = (sum > 0.f && __builtin_isfinite(sum)) ? 1.f / sum : 0.f;
        for (int c = 0; c < 256; ++c) rp[c] *= inv;
    }
    __syncthreads();
#pragma unroll
    for (int i = 0; i < 8; ++i) {            // P -> bf16, vectorized
        int e4 = (t + i * 256) * 4;
        float4 f = *(const float4*)&sbuf[e4];
        uint2 u; u.x = pk_bf16(f.x, f.y); u.y = pk_bf16(f.z, f.w);
        *(uint2*)&pbuf[e4] = u;
    }
    __syncthreads();

    // ---- phase 2: attended = P(32x256) @ V_sel(256x512); wave owns 64 cols --
    v8f accO[2][4];
#pragma unroll
    for (int rt = 0; rt < 2; ++rt)
#pragma unroll
        for (int cc = 0; cc < 4; ++cc) accO[rt][cc] = v8f_zero();

    for (int k0 = 0; k0 < KTOP; k0 += 32) {
#pragma unroll
        for (int i = 0; i < 4; ++i) {        // vs[c][k]: 4x4 register transpose
            int q  = t + i * 256;            // 1024 quads
            int k4 = (q >> 7) * 4, c4 = (q & 127) * 4;
            float4 r0 = *(const float4*)&vb[(long long)(k0 + k4 + 0) * C_SZ + c4];
            float4 r1 = *(const float4*)&vb[(long long)(k0 + k4 + 1) * C_SZ + c4];
            float4 r2 = *(const float4*)&vb[(long long)(k0 + k4 + 2) * C_SZ + c4];
            float4 r3 = *(const float4*)&vb[(long long)(k0 + k4 + 3) * C_SZ + c4];
            float a0[4] = {r0.x, r0.y, r0.z, r0.w};
            float a1[4] = {r1.x, r1.y, r1.z, r1.w};
            float a2[4] = {r2.x, r2.y, r2.z, r2.w};
            float a3[4] = {r3.x, r3.y, r3.z, r3.w};
#pragma unroll
            for (int j2 = 0; j2 < 4; ++j2) {
                uint2 u;
                u.x = pk_bf16(a0[j2], a1[j2]);
                u.y = pk_bf16(a2[j2], a3[j2]);
                *(uint2*)&vs[(c4 + j2) * 32 + k4] = u;
            }
        }
        __syncthreads();

        v16bf af[2], bfr[4];
#pragma unroll
        for (int rt = 0; rt < 2; ++rt) {
            const __bf16* p = &pbuf[(rt * 16 + lrow) * 256 + k0 + half * 8];
            af[rt] = frag_from_lds(p, p + 16);
        }
#pragma unroll
        for (int cc = 0; cc < 4; ++cc) {
            const __bf16* p = &vs[(wave * 64 + cc * 16 + lrow) * 32 + half * 8];
            bfr[cc] = frag_from_lds(p, p + 16);
        }
#pragma unroll
        for (int rt = 0; rt < 2; ++rt)
#pragma unroll
            for (int cc = 0; cc < 4; ++cc)
                accO[rt][cc] = wmma_bf16(af[rt], bfr[cc], accO[rt][cc]);
        __syncthreads();
    }

    // scatter attended into vbuf columns idx[j] (blocks own disjoint rows j)
    const long long vbase = (long long)b * C_SZ * N_SZ;
#pragma unroll
    for (int rt = 0; rt < 2; ++rt)
#pragma unroll
        for (int cc = 0; cc < 4; ++cc)
#pragma unroll
            for (int r = 0; r < 8; ++r) {
                int jl = j0 + rt * 16 + r + half * 8;
                int c  = wave * 64 + cc * 16 + lrow;
                vbuf[vbase + (long long)c * N_SZ + ib[jl]] = accO[rt][cc][r];
            }
}

// ---------------------------------------------------------------------------
// Host-side orchestration.  Workspace layout (bytes), total ~96.2 MB:
//   imp   @ 0          : B*N*4          = 147456
//   idx   @ 147456     : B*K*4          = 16384
//   vbuf  @ 163840     : B*C*N*4        = 75497472
//   qsel  @ 75661312   : B*K*C*4        = 8388608
//   ksel  @ 84049920   : B*K*C*4        = 8388608
//   vsel  @ 92438528   : B*K*C*4        = 8388608
// ---------------------------------------------------------------------------
extern "C" void kernel_launch(void* const* d_in, const int* in_sizes, int n_in,
                              void* d_out, int out_size, void* d_ws, size_t ws_size,
                              hipStream_t stream) {
    const float* x   = (const float*)d_in[0];
    const float* wq  = (const float*)d_in[1];
    const float* wk  = (const float*)d_in[2];
    const float* wv  = (const float*)d_in[3];
    const float* wo  = (const float*)d_in[4];
    const float* lnw = (const float*)d_in[5];
    const float* lnb = (const float*)d_in[6];
    float* out = (float*)d_out;

    char* ws = (char*)d_ws;
    float* imp  = (float*)(ws + 0);
    int*   idx  = (int*)  (ws + 147456);
    float* vbuf = (float*)(ws + 163840);
    float* qsel = (float*)(ws + 163840 + 75497472LL);
    float* ksel = (float*)(ws + 163840 + 75497472LL + 8388608LL);
    float* vsel = (float*)(ws + 163840 + 75497472LL + 2 * 8388608LL);

    const long long sCN = (long long)C_SZ * N_SZ;
    const long long sKC = (long long)KTOP * C_SZ;
    const float scale = 1.0f / sqrtf((float)C_SZ);

    importance_kernel<<<dim3((B_SZ * N_SZ) / 256), 256, 0, stream>>>(x, imp);
    topk_kernel<<<dim3(B_SZ), 256, 0, stream>>>(imp, idx);

    // v = wv @ x  (full, (b,c,n))
    gemm_wmma_kernel<<<dim3(N_SZ / 128, C_SZ / 128, B_SZ), 256, 0, stream>>>(
        wv, x, sCN, vbuf, sCN, nullptr, 0, nullptr, 0,
        C_SZ, C_SZ, N_SZ, N_SZ, 0);
    // q_sel = (wq @ x)[:, idx]  stored (b,j,c)
    gemm_wmma_kernel<<<dim3(KTOP / 128, C_SZ / 128, B_SZ), 256, 0, stream>>>(
        wq, x, sCN, qsel, sKC, idx, KTOP, nullptr, 0,
        C_SZ, C_SZ, KTOP, N_SZ, 1);
    gemm_wmma_kernel<<<dim3(KTOP / 128, C_SZ / 128, B_SZ), 256, 0, stream>>>(
        wk, x, sCN, ksel, sKC, idx, KTOP, nullptr, 0,
        C_SZ, C_SZ, KTOP, N_SZ, 1);

    layernorm_kernel<<<dim3(B_SZ * KTOP), 256, 0, stream>>>(qsel, lnw, lnb);
    layernorm_kernel<<<dim3(B_SZ * KTOP), 256, 0, stream>>>(ksel, lnw, lnb);

    gather_vsel_kernel<<<dim3((B_SZ * KTOP * C_SZ) / 256), 256, 0, stream>>>(
        vbuf, idx, vsel);

    attn_wmma_kernel<<<dim3(KTOP / 32, B_SZ), 256, 0, stream>>>(
        qsel, ksel, vsel, idx, vbuf, scale);

    // y = x + wo @ vbuf
    gemm_wmma_kernel<<<dim3(N_SZ / 128, C_SZ / 128, B_SZ), 256, 0, stream>>>(
        wo, vbuf, sCN, out, sCN, nullptr, 0, x, sCN,
        C_SZ, C_SZ, N_SZ, N_SZ, 0);

    (void)in_sizes; (void)n_in; (void)out_size; (void)ws_size;
}
